// Gate_28363964023126
// MI455X (gfx1250) — compile-verified, matrix operand order checked by reading, not run on hardware
//
#include <hip/hip_runtime.h>

#define DIM        7168
#define N_EXPERTS  256
#define N_GROUPS   8
#define GROUP_SZ   32
#define TOPK_GRPS  4
#define TOPK       8
#define ROUTE_SCALE 2.5f

#define TOK_PER_BLK 64          // M per block
#define M_TILES     4           // 4 x 16-token WMMA tiles
#define K_CHUNK     128         // K staged to LDS per round
#define K_STEP      32          // bf16 WMMA K
#define A_STRIDE    136         // ushorts per LDS row (128 + 8 pad) -> 68 dwords, 68%64=4
#define SC_STRIDE   264         // floats per score row

typedef __attribute__((ext_vector_type(16))) __bf16 v16bf;
typedef __attribute__((ext_vector_type(8)))  float  v8f;

union FragBF { v16bf v; unsigned int u[8]; uint4 q[2]; };

// two fp32 -> packed bf16 pair (truncation), single v_perm_b32
__device__ __forceinline__ unsigned int pk_bf16(float lo, float hi) {
    return __builtin_amdgcn_perm(__float_as_uint(hi), __float_as_uint(lo), 0x07060302u);
}

// ---------------- Pass 1: W fp32 -> bf16 (once, lives in L2) ----------------
__global__ __launch_bounds__(256) void convert_w_kernel(const float* __restrict__ w,
                                                        unsigned short* __restrict__ wb) {
    const int idx = blockIdx.x * 256 + threadIdx.x;   // float4 index
    const float4 f = reinterpret_cast<const float4*>(w)[idx];
    uint2 o;
    o.x = pk_bf16(f.x, f.y);
    o.y = pk_bf16(f.z, f.w);
    reinterpret_cast<uint2*>(wb)[idx] = o;
}

// ---------------- Pass 2: fused scores GEMM + routing ----------------
__global__ __launch_bounds__(256) void
moe_gate_kernel(const float* __restrict__ x,
                const unsigned short* __restrict__ wbf,   // [256][7168] bf16
                const float* __restrict__ expert_bias,
                float* __restrict__ out_w,
                int*   __restrict__ out_i,
                int n_tokens)
{
    extern __shared__ unsigned char smem[];
    unsigned short* a_sh = reinterpret_cast<unsigned short*>(smem);          // [64][A_STRIDE] bf16
    float*          sc   = reinterpret_cast<float*>(smem + TOK_PER_BLK * A_STRIDE * 2); // [64][SC_STRIDE]
    __shared__ float bsh[N_EXPERTS];

    const int tid  = threadIdx.x;
    const int wave = tid >> 5;
    const int lane = tid & 31;
    const int m0   = blockIdx.x * TOK_PER_BLK;

    bsh[tid] = expert_bias[tid];              // blockDim == 256 == N_EXPERTS

    const int r      = lane & 15;             // M within tile for A / N for B
    const int khalf  = lane >> 4;             // 0: K{0..7,16..23}, 1: K{8..15,24..31}
    const int e_base = wave * 32;

    const unsigned short* __restrict__ wr0 = wbf + (size_t)(e_base + r)      * DIM;
    const unsigned short* __restrict__ wr1 = wbf + (size_t)(e_base + 16 + r) * DIM;

    // staging thread mapping: 64 rows x 32 float4 per K_CHUNK, 8 float4/thread
    const int srow = (tid * 8) >> 5 >> 3;     // placeholder (computed per-i below)
    (void)srow;

    v8f acc[M_TILES][2];
    #pragma unroll
    for (int mt = 0; mt < M_TILES; ++mt) { acc[mt][0] = (v8f){}; acc[mt][1] = (v8f){}; }

    const int NCH = DIM / K_CHUNK;            // 56 chunks

    float4 G[8];
    // prologue: load chunk 0
    {
        #pragma unroll
        for (int i = 0; i < 8; ++i) {
            const int idx = i * 256 + tid;    // float4 index in 64x128 chunk
            const int row = idx >> 5;         // 32 float4 per row
            const int c4  = idx & 31;
            G[i] = reinterpret_cast<const float4*>(x + (size_t)(m0 + row) * DIM)[c4];
        }
    }

    for (int c = 0; c < NCH; ++c) {
        const int k0 = c * K_CHUNK;

        __syncthreads();                      // previous GEMM done reading a_sh
        // convert + store staged chunk
        #pragma unroll
        for (int i = 0; i < 8; ++i) {
            const int idx = i * 256 + tid;
            const int row = idx >> 5;
            const int c4  = idx & 31;
            uint2 o;
            o.x = pk_bf16(G[i].x, G[i].y);
            o.y = pk_bf16(G[i].z, G[i].w);
            *reinterpret_cast<uint2*>(a_sh + row * A_STRIDE + c4 * 4) = o;
        }
        __syncthreads();

        // prefetch next chunk while GEMM runs
        if (c + 1 < NCH) {
            const int kn = k0 + K_CHUNK;
            #pragma unroll
            for (int i = 0; i < 8; ++i) {
                const int idx = i * 256 + tid;
                const int row = idx >> 5;
                const int c4  = idx & 31;
                G[i] = reinterpret_cast<const float4*>(x + (size_t)(m0 + row) * DIM + kn)[c4];
            }
        }

        // GEMM over this chunk: 4 k-steps x (2 B-tiles x 4 M-tiles)
        #pragma unroll
        for (int ks = 0; ks < K_CHUNK / K_STEP; ++ks) {
            const int kk = ks * K_STEP;
            const int o0 = kk + khalf * 8;        // local bf16 element offset
            const int go = k0 + o0;               // global bf16 element offset (B)

            FragBF b0, b1;
            b0.q[0] = *reinterpret_cast<const uint4*>(wr0 + go);
            b0.q[1] = *reinterpret_cast<const uint4*>(wr0 + go + 16);
            b1.q[0] = *reinterpret_cast<const uint4*>(wr1 + go);
            b1.q[1] = *reinterpret_cast<const uint4*>(wr1 + go + 16);

            #pragma unroll
            for (int mt = 0; mt < M_TILES; ++mt) {
                const unsigned short* ar = a_sh + (mt * 16 + r) * A_STRIDE + o0;
                FragBF a;
                a.q[0] = *reinterpret_cast<const uint4*>(ar);
                a.q[1] = *reinterpret_cast<const uint4*>(ar + 16);
                acc[mt][0] = __builtin_amdgcn_wmma_f32_16x16x32_bf16(
                                 false, a.v, false, b0.v, (short)0, acc[mt][0], false, false);
                acc[mt][1] = __builtin_amdgcn_wmma_f32_16x16x32_bf16(
                                 false, a.v, false, b1.v, (short)0, acc[mt][1], false, false);
            }
        }
    }

    // sigmoid + scatter: D layout: VGPR j, lane L -> (M = j + 8*(L>>4), N = L&15)
    const int Mofs = khalf * 8;
    const int nc0  = e_base + r;
    const int nc1  = e_base + 16 + r;
    #pragma unroll
    for (int mt = 0; mt < M_TILES; ++mt) {
        #pragma unroll
        for (int j = 0; j < 8; ++j) {
            const int row = mt * 16 + Mofs + j;
            sc[row * SC_STRIDE + nc0] = 1.0f / (1.0f + __expf(-acc[mt][0][j]));
            sc[row * SC_STRIDE + nc1] = 1.0f / (1.0f + __expf(-acc[mt][1][j]));
        }
    }

    __syncthreads();

    // -------- Routing: one thread per token (64 tokens) --------
    if (tid < TOK_PER_BLK) {
        const int t     = tid;
        const int token = m0 + t;
        const float* srow_p = sc + t * SC_STRIDE;

        float gscore[N_GROUPS];
        #pragma unroll
        for (int g = 0; g < N_GROUPS; ++g) {
            float m1 = -__builtin_inff(), m2 = -__builtin_inff();
            for (int j = 0; j < GROUP_SZ; ++j) {
                const int e = g * GROUP_SZ + j;
                const float v = srow_p[e] + bsh[e];
                if (v > m1)      { m2 = m1; m1 = v; }
                else if (v > m2) { m2 = v; }
            }
            gscore[g] = m1 + m2;
        }

        bool keep[N_GROUPS];
        #pragma unroll
        for (int g = 0; g < N_GROUPS; ++g) keep[g] = false;
        for (int it = 0; it < TOPK_GRPS; ++it) {
            float best = -__builtin_inff(); int bi = 0;
            for (int g = 0; g < N_GROUPS; ++g)
                if (!keep[g] && gscore[g] > best) { best = gscore[g]; bi = g; }
            keep[bi] = true;
        }

        float tv[TOPK]; int ti[TOPK];
        #pragma unroll
        for (int i = 0; i < TOPK; ++i) { tv[i] = -__builtin_inff(); ti[i] = -1; }
        for (int g = 0; g < N_GROUPS; ++g) {
            if (!keep[g]) continue;
            for (int j = 0; j < GROUP_SZ; ++j) {
                const int e = g * GROUP_SZ + j;
                const float v = srow_p[e] + bsh[e];
                if (v > tv[TOPK - 1]) {
                    int p = TOPK - 1;
                    while (p > 0 && tv[p - 1] < v) {
                        tv[p] = tv[p - 1]; ti[p] = ti[p - 1]; --p;
                    }
                    tv[p] = v; ti[p] = e;
                }
            }
        }

        float wsum = 0.0f, wv[TOPK];
        #pragma unroll
        for (int i = 0; i < TOPK; ++i) { wv[i] = srow_p[ti[i]]; wsum += wv[i]; }
        const float norm = ROUTE_SCALE / (wsum + 1e-20f);
        #pragma unroll
        for (int i = 0; i < TOPK; ++i) {
            out_w[(size_t)token * TOPK + i] = wv[i] * norm;
            out_i[(size_t)token * TOPK + i] = ti[i];
        }
    }
}

extern "C" void kernel_launch(void* const* d_in, const int* in_sizes, int n_in,
                              void* d_out, int out_size, void* d_ws, size_t ws_size,
                              hipStream_t stream) {
    const float* x           = (const float*)d_in[0];
    const float* weight      = (const float*)d_in[1];
    const float* expert_bias = (const float*)d_in[2];

    const int n_tokens = in_sizes[0] / DIM;

    unsigned short* wbf = (unsigned short*)d_ws;      // 256*7168*2 = 3.67 MB scratch

    float* out_w = (float*)d_out;                                   // [n, 8] f32
    int*   out_i = (int*)((float*)d_out + (size_t)n_tokens * TOPK); // [n, 8] i32 (bitwise)

    // Pass 1: W -> bf16
    convert_w_kernel<<<(N_EXPERTS * DIM / 4) / 256, 256, 0, stream>>>(weight, wbf);

    // Pass 2: fused GEMM + routing
    const int blocks = n_tokens / TOK_PER_BLK;
    const size_t smem_bytes = (size_t)TOK_PER_BLK * A_STRIDE * 2      // A tile (bf16)
                            + (size_t)TOK_PER_BLK * SC_STRIDE * 4;    // score tile (f32)
    moe_gate_kernel<<<blocks, 256, smem_bytes, stream>>>(x, wbf, expert_bias,
                                                         out_w, out_i, n_tokens);
}
